// SimpleNN_4355096838716
// MI455X (gfx1250) — compile-verified, matrix operand order checked by reading, not run on hardware
//
#include <hip/hip_runtime.h>
#include <hip/hip_bf16.h>
#include <limits.h>

typedef __attribute__((ext_vector_type(16))) _Float16 v16h;
typedef __attribute__((ext_vector_type(8)))  _Float16 v8h;
typedef __attribute__((ext_vector_type(8)))  float    v8f;
typedef __attribute__((ext_vector_type(2)))  float    v2f;
typedef __attribute__((ext_vector_type(4)))  int      v4i;

#define DEV static __device__ __forceinline__

// Order-preserving float<->int encoding so integer atomicMax == float max.
DEV int   encf(float f){ int i = __float_as_int(f); return (i >= 0) ? i : (i ^ 0x7fffffff); }
DEV float decf(int i){ return __int_as_float((i >= 0) ? i : (i ^ 0x7fffffff)); }

#if defined(__has_builtin)
#if __has_builtin(__builtin_amdgcn_wmma_f32_16x16x4_f32)
#define HAVE_WMMA_F32K4 1
#endif
#if __has_builtin(__builtin_amdgcn_global_load_async_to_lds_b128) && \
    __has_builtin(__builtin_amdgcn_s_wait_asynccnt)
#define HAVE_ASYNC_LDS 1
#endif
#endif

// ----------------------------------------------------------------------------
// init buffers
// ----------------------------------------------------------------------------
__global__ __launch_bounds__(256) void k_init_int(int* buf, int count, int val) {
  int i = blockIdx.x * 256 + threadIdx.x;
  if (i < count) buf[i] = val;
}

// ----------------------------------------------------------------------------
// Prep layer-1: pack 4x[3,64] weights and 4x[64] biases into single buffers.
// ----------------------------------------------------------------------------
__global__ __launch_bounds__(256) void k_prep_l1(
    const float* __restrict__ Wq, const float* __restrict__ Wk,
    const float* __restrict__ Wv, const float* __restrict__ Ws,
    const float* __restrict__ bq, const float* __restrict__ bk,
    const float* __restrict__ bv, const float* __restrict__ bs,
    float* __restrict__ wcat, float* __restrict__ bcat) {
  int idx = blockIdx.x * 256 + threadIdx.x;
  if (idx < 4 * 64) {
    int m = idx >> 6, c = idx & 63;
    const float* B = (m == 0) ? bq : (m == 1) ? bk : (m == 2) ? bv : bs;
    bcat[idx] = B[c];
  }
  if (idx >= 4 * 192) return;
  int m = idx / 192, r = idx % 192;
  const float* W = (m == 0) ? Wq : (m == 1) ? Wk : (m == 2) ? Wv : Ws;
  wcat[idx] = W[r];
}

// ----------------------------------------------------------------------------
// Prep layer-2: convert 4x[64,128] f32 weights into f16 *fragment order* so a
// lane's whole v16h B-fragment is one contiguous 32B read, + pack biases.
// Fragment layout: frag = m*16 + kt*8 + nt; element = lane*16 + i.
// Within a lane (per ISA 7.12.2): i<8 -> K = kt*32 + kbase + i,
//                                 i>=8 -> K = kt*32 + 16 + kbase + (i-8),
// kbase = (lane<16) ? 0 : 8, col = lane&15, ncol = nt*16+col.
// ----------------------------------------------------------------------------
__global__ __launch_bounds__(256) void k_prep_l2(
    const float* __restrict__ Wq, const float* __restrict__ Wk,
    const float* __restrict__ Wv, const float* __restrict__ Ws,
    const float* __restrict__ bq, const float* __restrict__ bk,
    const float* __restrict__ bv, const float* __restrict__ bs,
    _Float16* __restrict__ wfrag, float* __restrict__ bcat) {
  int idx = blockIdx.x * 256 + threadIdx.x;
  if (idx < 4 * 128) {
    int m = idx >> 7, c = idx & 127;
    const float* B = (m == 0) ? bq : (m == 1) ? bk : (m == 2) ? bv : bs;
    bcat[idx] = B[c];
  }
  if (idx >= 4 * 2 * 8 * 512) return;
  int frag = idx >> 9;                       // 0..63
  int m = frag >> 4, kt = (frag >> 3) & 1, nt = frag & 7;
  int r = idx & 511;
  int lane = r >> 4, i = r & 15;
  int kbase = (lane < 16) ? 0 : 8;
  int col = lane & 15;
  int K = kt * 32 + ((i < 8) ? (kbase + i) : (16 + kbase + (i - 8)));
  int ncol = nt * 16 + col;
  const float* W = (m == 0) ? Wq : (m == 1) ? Wk : (m == 2) ? Wv : Ws;
  wfrag[idx] = (_Float16)W[K * 128 + ncol];
}

// ----------------------------------------------------------------------------
// Layer-1 projections: [N,3] @ [3,64] (+bias), q/k/v/skip fused into qkvs1.
// V_WMMA_F32_16X16X4_F32 (K padded 3->4); scalar fallback if builtin absent.
// ----------------------------------------------------------------------------
#ifdef HAVE_WMMA_F32K4
__global__ __launch_bounds__(256) void k_l1_proj_wmma(
    const float* __restrict__ x, const float* __restrict__ wcat,
    const float* __restrict__ bcat, float* __restrict__ qkvs1, int n) {
  const int rowbase = blockIdx.x * 16;
  const int t = threadIdx.x, wave = t >> 5, lane = t & 31;
  const int col = lane & 15;
  const int k0  = (lane < 16) ? 0 : 2;
  const int arow = rowbase + col;
  const int arow_c = (arow < n) ? arow : (n - 1);

  v2f a;
  a[0] = (k0 + 0 < 3) ? x[(size_t)arow_c * 3 + k0 + 0] : 0.0f;
  a[1] = (k0 + 1 < 3) ? x[(size_t)arow_c * 3 + k0 + 1] : 0.0f;

  const bool full = (rowbase + 16 <= n);
  const int rofs = (lane < 16) ? 0 : 8;

#pragma unroll
  for (int tt = 0; tt < 2; ++tt) {
    int tile = wave * 2 + tt;           // 0..15
    int m = tile >> 2, nt = tile & 3;
    int ncol = nt * 16 + col;
    v2f b;
    b[0] = (k0 + 0 < 3) ? wcat[m * 192 + (k0 + 0) * 64 + ncol] : 0.0f;
    b[1] = (k0 + 1 < 3) ? wcat[m * 192 + (k0 + 1) * 64 + ncol] : 0.0f;
    v8f acc;
    float bias = bcat[m * 64 + ncol];
#pragma unroll
    for (int r = 0; r < 8; ++r) acc[r] = bias;
    acc = __builtin_amdgcn_wmma_f32_16x16x4_f32(false, a, false, b,
                                                (short)0, acc, false, false);
    float* O = qkvs1 + (size_t)m * n * 64;
    if (full) {
#pragma unroll
      for (int r = 0; r < 8; ++r)
        O[(size_t)(rowbase + rofs + r) * 64 + ncol] = acc[r];
    } else {
#pragma unroll
      for (int r = 0; r < 8; ++r) {
        int row = rowbase + rofs + r;
        if (row < n) O[(size_t)row * 64 + ncol] = acc[r];
      }
    }
  }
}
#else
__global__ __launch_bounds__(256) void k_l1_proj_scalar(
    const float* __restrict__ x, const float* __restrict__ wcat,
    const float* __restrict__ bcat, float* __restrict__ qkvs1, int n) {
  int idx = blockIdx.x * 256 + threadIdx.x;
  if (idx >= n * 64) return;
  int row = idx >> 6, c = idx & 63;
  float x0 = x[(size_t)row * 3 + 0], x1 = x[(size_t)row * 3 + 1],
        x2 = x[(size_t)row * 3 + 2];
#pragma unroll
  for (int m = 0; m < 4; ++m) {
    const float* W = wcat + m * 192;
    qkvs1[(size_t)m * n * 64 + idx] =
        bcat[m * 64 + c] + x0 * W[c] + x1 * W[64 + c] + x2 * W[128 + c];
  }
}
#endif

// ----------------------------------------------------------------------------
// Layer-2 projections: [N,64] f16 @ 4x[64,128] f16 -> f32 (fused q/k/v/skip).
// A tile async-staged to LDS (padded stride 80 halves), B read as contiguous
// pre-swizzled fragments. 16-row M-tile per block; wave w owns N-tile w.
// ----------------------------------------------------------------------------
__global__ __launch_bounds__(256) void k_l2_proj_wmma(
    const _Float16* __restrict__ h1,     // [N,64]
    const _Float16* __restrict__ wfrag,  // fragment-ordered weights
    const float* __restrict__ bcat,      // [4][128]
    float* __restrict__ qkvs2, int n) {
  __shared__ _Float16 As[16 * 80];       // 80-half row stride (16B aligned)
  const int rowbase = blockIdx.x * 16;
  const int t = threadIdx.x, wave = t >> 5, lane = t & 31;

#ifdef HAVE_ASYNC_LDS
  if (t < 128) {                         // 128 x b128 = 16 rows x 64 halves
    int r = t >> 3, c8 = (t & 7) * 8;
    int row = rowbase + r;
    row = (row < n) ? row : (n - 1);
    __builtin_amdgcn_global_load_async_to_lds_b128(
        (__attribute__((address_space(1))) v4i*)(h1 + (size_t)row * 64 + c8),
        (__attribute__((address_space(3))) v4i*)&As[r * 80 + c8], 0, 0);
  }
  __builtin_amdgcn_s_wait_asynccnt(0);
  __syncthreads();
#else
  {
    int i = t * 4;                        // 4 contiguous halves per thread
    int r = i >> 6, c = i & 63;
    int row = rowbase + r;
    row = (row < n) ? row : (n - 1);
    *(uint2*)&As[r * 80 + c] = *(const uint2*)&h1[(size_t)row * 64 + c];
  }
  __syncthreads();
#endif

  const int col   = lane & 15;
  const int kbase = (lane < 16) ? 0 : 8;
  const int nt    = wave;                // N-tile 0..7
  const bool full = (rowbase + 16 <= n);
  const int rofs  = (lane < 16) ? 0 : 8;

#pragma unroll
  for (int m = 0; m < 4; ++m) {
    v8f acc;
    float bias = bcat[m * 128 + nt * 16 + col];
#pragma unroll
    for (int r = 0; r < 8; ++r) acc[r] = bias;
#pragma unroll
    for (int kt = 0; kt < 2; ++kt) {
      // A fragment: two contiguous 16B chunks from LDS.
      const v8h lo = *(const v8h*)&As[col * 80 + kt * 32 + kbase];
      const v8h hi = *(const v8h*)&As[col * 80 + kt * 32 + 16 + kbase];
      v16h afrag;
#pragma unroll
      for (int i = 0; i < 8; ++i) { afrag[i] = lo[i]; afrag[8 + i] = hi[i]; }
      // B fragment: one contiguous 32B read (pre-swizzled).
      const v16h bfrag =
          *(const v16h*)&wfrag[((size_t)(m * 16 + kt * 8 + nt) * 512) + lane * 16];
      acc = __builtin_amdgcn_wmma_f32_16x16x32_f16(false, afrag, false, bfrag,
                                                   (short)0, acc, false, false);
    }
    float* O = qkvs2 + (size_t)m * n * 128;
    int ncol = nt * 16 + col;
    if (full) {
#pragma unroll
      for (int r = 0; r < 8; ++r)
        O[(size_t)(rowbase + rofs + r) * 128 + ncol] = acc[r];
    } else {
#pragma unroll
      for (int r = 0; r < 8; ++r) {
        int row = rowbase + rofs + r;
        if (row < n) O[(size_t)row * 128 + ncol] = acc[r];
      }
    }
  }
}

// ----------------------------------------------------------------------------
// Edge pass 1 (one wave per edge): attention logits + segment max.
// D = per-head dim, channels = 2*D.
// ----------------------------------------------------------------------------
template <int D>
__global__ __launch_bounds__(256) void k_edge_logits(
    const int* __restrict__ ei, const float* __restrict__ q,
    const float* __restrict__ k, float* __restrict__ logit,
    int* __restrict__ mbuf, int E) {
  int e = blockIdx.x * 8 + (threadIdx.x >> 5);
  if (e >= E) return;
  int lane = threadIdx.x & 31;
  int src = ei[e], dst = ei[E + e];
  const float* qd = q + (size_t)dst * 2 * D;
  const float* ks = k + (size_t)src * 2 * D;
  float p0 = 0.0f, p1 = 0.0f;
#pragma unroll
  for (int j = lane; j < D; j += 32) {
    p0 += qd[j] * ks[j];
    p1 += qd[D + j] * ks[D + j];
  }
#pragma unroll
  for (int o = 16; o > 0; o >>= 1) {
    p0 += __shfl_xor(p0, o, 32);
    p1 += __shfl_xor(p1, o, 32);
  }
  if (lane == 0) {
    float inv = rsqrtf((float)D);
    float s0 = p0 * inv, s1 = p1 * inv;
    logit[(size_t)e * 2 + 0] = s0;
    logit[(size_t)e * 2 + 1] = s1;
    atomicMax(&mbuf[dst * 2 + 0], encf(s0));
    atomicMax(&mbuf[dst * 2 + 1], encf(s1));
  }
}

// ----------------------------------------------------------------------------
// Edge pass 2 (one wave per edge): e = exp(logit - max), accumulate denom and
// the UNNORMALIZED weighted values; the e/denom division happens in finalize
// (sum(e*v)/denom == sum((e/denom)*v)).
// ----------------------------------------------------------------------------
template <int D>
__global__ __launch_bounds__(256) void k_edge_scatter(
    const int* __restrict__ ei, const float* __restrict__ logit,
    const int* __restrict__ mbuf, const float* __restrict__ v,
    float* __restrict__ denom, float* __restrict__ agg, int E) {
  int e = blockIdx.x * 8 + (threadIdx.x >> 5);
  if (e >= E) return;
  int lane = threadIdx.x & 31;
  int src = ei[e], dst = ei[E + e];
  float e0 = __expf(logit[(size_t)e * 2 + 0] - decf(mbuf[dst * 2 + 0]));
  float e1 = __expf(logit[(size_t)e * 2 + 1] - decf(mbuf[dst * 2 + 1]));
  if (lane == 0) {
    atomicAdd(&denom[dst * 2 + 0], e0);
    atomicAdd(&denom[dst * 2 + 1], e1);
  }
  const float* vs = v + (size_t)src * 2 * D;
  float* ad = agg + (size_t)dst * 2 * D;
#pragma unroll
  for (int j = lane; j < D; j += 32) {
    atomicAdd(&ad[j],     e0 * vs[j]);
    atomicAdd(&ad[D + j], e1 * vs[D + j]);
  }
}

// ----------------------------------------------------------------------------
// finalize: relu(agg/denom + skip); layer-1 also emits f16 for the WMMA GEMM.
// CH = total channels (2 heads); head = channel / (CH/2).
// ----------------------------------------------------------------------------
template <int CH>
__global__ __launch_bounds__(256) void k_finalize_f16(
    const float* __restrict__ agg, const float* __restrict__ denom,
    const float* __restrict__ skip, _Float16* __restrict__ out, int total) {
  int i = blockIdx.x * 256 + threadIdx.x;
  if (i >= total) return;
  int node = i / CH, c = i % CH;
  int h = (c >= CH / 2) ? 1 : 0;
  float d = fmaxf(denom[node * 2 + h], 1e-16f);
  out[i] = (_Float16)fmaxf(agg[i] / d + skip[i], 0.0f);
}

template <int CH>
__global__ __launch_bounds__(256) void k_finalize_f32(
    const float* __restrict__ agg, const float* __restrict__ denom,
    const float* __restrict__ skip, float* __restrict__ out, int total) {
  int i = blockIdx.x * 256 + threadIdx.x;
  if (i >= total) return;
  int node = i / CH, c = i % CH;
  int h = (c >= CH / 2) ? 1 : 0;
  float d = fmaxf(denom[node * 2 + h], 1e-16f);
  out[i] = fmaxf(agg[i] / d + skip[i], 0.0f);
}

// ----------------------------------------------------------------------------
// global max pool over graphs + decode
// ----------------------------------------------------------------------------
__global__ __launch_bounds__(256) void k_pool_max(
    const float* __restrict__ h2, const int* __restrict__ batch,
    int* __restrict__ pint, int n) {
  int idx = blockIdx.x * 256 + threadIdx.x;   // (node, channel)
  if (idx >= n * 128) return;
  int node = idx >> 7, c = idx & 127;
  atomicMax(&pint[batch[node] * 128 + c], encf(h2[idx]));
}

__global__ __launch_bounds__(256) void k_pool_decode(
    const int* __restrict__ pint, float* __restrict__ p, int total) {
  int i = blockIdx.x * 256 + threadIdx.x;
  if (i >= total) return;
  int v = pint[i];
  p[i] = (v == INT_MIN) ? 0.0f : decf(v);
}

// ----------------------------------------------------------------------------
// MLP head (tiny: G=64)
// ----------------------------------------------------------------------------
__global__ __launch_bounds__(256) void k_mlp1(
    const float* __restrict__ p, const float* __restrict__ W1,
    const float* __restrict__ b1, float* __restrict__ latent,
    float* __restrict__ out_latent) {
  int idx = blockIdx.x * 256 + threadIdx.x;   // 64*32
  if (idx >= 64 * 32) return;
  int g = idx >> 5, j = idx & 31;
  float s = b1[j];
  for (int c = 0; c < 128; ++c) s += p[g * 128 + c] * W1[c * 32 + j];
  s = fmaxf(s, 0.0f);
  latent[idx] = s;
  out_latent[idx] = s;
}

__global__ __launch_bounds__(256) void k_mlp2(
    const float* __restrict__ latent, const float* __restrict__ W2,
    const float* __restrict__ b2, float* __restrict__ y) {
  int idx = blockIdx.x * 256 + threadIdx.x;   // 64*128
  if (idx >= 64 * 128) return;
  int g = idx >> 7, j = idx & 127;
  float s = b2[j];
  for (int c = 0; c < 32; ++c) s += latent[g * 32 + c] * W2[c * 128 + j];
  y[idx] = fmaxf(s, 0.0f);
}

__global__ __launch_bounds__(256) void k_mlp3(
    const float* __restrict__ y, const float* __restrict__ W3,
    const float* __restrict__ b3, float* __restrict__ out) {
  int idx = blockIdx.x * 256 + threadIdx.x;   // 64*40
  if (idx >= 64 * 40) return;
  int g = idx / 40, j = idx % 40;
  float s = b3[j];
  for (int c = 0; c < 128; ++c) s += y[g * 128 + c] * W3[c * 40 + j];
  out[idx] = s;
}

// ----------------------------------------------------------------------------
// host launcher
// ----------------------------------------------------------------------------
extern "C" void kernel_launch(void* const* d_in, const int* in_sizes, int n_in,
                              void* d_out, int out_size, void* d_ws, size_t ws_size,
                              hipStream_t stream) {
  const float* x     = (const float*)d_in[0];
  const int*   ei    = (const int*)d_in[1];
  const int*   batch = (const int*)d_in[2];
  const float *Wq1 = (const float*)d_in[3],  *bq1 = (const float*)d_in[4];
  const float *Wk1 = (const float*)d_in[5],  *bk1 = (const float*)d_in[6];
  const float *Wv1 = (const float*)d_in[7],  *bv1 = (const float*)d_in[8];
  const float *Ws1 = (const float*)d_in[9],  *bs1 = (const float*)d_in[10];
  const float *Wq2 = (const float*)d_in[11], *bq2 = (const float*)d_in[12];
  const float *Wk2 = (const float*)d_in[13], *bk2 = (const float*)d_in[14];
  const float *Wv2 = (const float*)d_in[15], *bv2 = (const float*)d_in[16];
  const float *Ws2 = (const float*)d_in[17], *bs2 = (const float*)d_in[18];
  const float *W1  = (const float*)d_in[19], *b1  = (const float*)d_in[20];
  const float *W2  = (const float*)d_in[21], *b2  = (const float*)d_in[22];
  const float *W3  = (const float*)d_in[23], *b3  = (const float*)d_in[24];

  const int N = in_sizes[0] / 3;
  const int E = in_sizes[1] / 2;
  const int G = 64;

  // ---- carve workspace ----
  char* cur = (char*)d_ws;
  auto carve = [&](size_t bytes) -> char* {
    char* p = cur;
    cur += (bytes + 255) & ~(size_t)255;
    return p;
  };
  float*    wcat1 = (float*)carve((size_t)4 * 192 * 4);
  float*    bcat1 = (float*)carve((size_t)4 * 64 * 4);
  _Float16* wfrag2 = (_Float16*)carve((size_t)4 * 2 * 8 * 512 * 2);
  float*    bcat2 = (float*)carve((size_t)4 * 128 * 4);
  float*    qkvs1 = (float*)carve((size_t)4 * N * 64 * 4);
  float*    agg1  = (float*)carve((size_t)N * 64 * 4);
  _Float16* h1h   = (_Float16*)carve((size_t)N * 64 * 2);
  float*    qkvs2 = (float*)carve((size_t)4 * N * 128 * 4);
  float*    agg2  = (float*)carve((size_t)N * 128 * 4);
  float*    h2    = (float*)carve((size_t)N * 128 * 4);
  float*    logit = (float*)carve((size_t)E * 2 * 4);
  int*      m1    = (int*)carve((size_t)N * 2 * 4);
  int*      m2    = (int*)carve((size_t)N * 2 * 4);
  float*    dn1   = (float*)carve((size_t)N * 2 * 4);
  float*    dn2   = (float*)carve((size_t)N * 2 * 4);
  int*      pint  = (int*)carve((size_t)G * 128 * 4);
  float*    pf    = (float*)carve((size_t)G * 128 * 4);
  float*    lat   = (float*)carve((size_t)G * 32 * 4);
  float*    ymid  = (float*)carve((size_t)G * 128 * 4);

  float* q1 = qkvs1;
  float* k1 = qkvs1 + (size_t)N * 64;
  float* v1 = qkvs1 + (size_t)2 * N * 64;
  float* s1 = qkvs1 + (size_t)3 * N * 64;
  float* q2 = qkvs2;
  float* k2 = qkvs2 + (size_t)N * 128;
  float* v2 = qkvs2 + (size_t)2 * N * 128;
  float* s2 = qkvs2 + (size_t)3 * N * 128;

  // ---- per-call init (deterministic, capture-safe) ----
  (void)hipMemsetAsync(agg1, 0, (size_t)N * 64 * 4, stream);
  (void)hipMemsetAsync(agg2, 0, (size_t)N * 128 * 4, stream);
  (void)hipMemsetAsync(dn1, 0, (size_t)N * 2 * 4, stream);
  (void)hipMemsetAsync(dn2, 0, (size_t)N * 2 * 4, stream);
  k_init_int<<<(N * 2 + 255) / 256, 256, 0, stream>>>(m1, N * 2, INT_MIN);
  k_init_int<<<(N * 2 + 255) / 256, 256, 0, stream>>>(m2, N * 2, INT_MIN);
  k_init_int<<<(G * 128 + 255) / 256, 256, 0, stream>>>(pint, G * 128, INT_MIN);

  // ---- weight prep ----
  k_prep_l1<<<3, 256, 0, stream>>>(Wq1, Wk1, Wv1, Ws1, bq1, bk1, bv1, bs1,
                                   wcat1, bcat1);
  k_prep_l2<<<128, 256, 0, stream>>>(Wq2, Wk2, Wv2, Ws2, bq2, bk2, bv2, bs2,
                                     wfrag2, bcat2);

  const int nb16 = (N + 15) / 16;

  // ---- layer 1 ----
#ifdef HAVE_WMMA_F32K4
  k_l1_proj_wmma<<<nb16, 256, 0, stream>>>(x, wcat1, bcat1, qkvs1, N);
#else
  k_l1_proj_scalar<<<(N * 64 + 255) / 256, 256, 0, stream>>>(x, wcat1, bcat1,
                                                             qkvs1, N);
#endif
  k_edge_logits<32><<<(E + 7) / 8, 256, 0, stream>>>(ei, q1, k1, logit, m1, E);
  k_edge_scatter<32><<<(E + 7) / 8, 256, 0, stream>>>(ei, logit, m1, v1, dn1,
                                                      agg1, E);
  k_finalize_f16<64><<<(N * 64 + 255) / 256, 256, 0, stream>>>(agg1, dn1, s1,
                                                               h1h, N * 64);

  // ---- layer 2 ----
  k_l2_proj_wmma<<<nb16, 256, 0, stream>>>(h1h, wfrag2, bcat2, qkvs2, N);
  k_edge_logits<64><<<(E + 7) / 8, 256, 0, stream>>>(ei, q2, k2, logit, m2, E);
  k_edge_scatter<64><<<(E + 7) / 8, 256, 0, stream>>>(ei, logit, m2, v2, dn2,
                                                      agg2, E);
  k_finalize_f32<128><<<(N * 128 + 255) / 256, 256, 0, stream>>>(agg2, dn2, s2,
                                                                 h2, N * 128);

  // ---- pool + MLP head ----
  k_pool_max<<<(N * 128 + 255) / 256, 256, 0, stream>>>(h2, batch, pint, N);
  k_pool_decode<<<(G * 128 + 255) / 256, 256, 0, stream>>>(pint, pf, G * 128);

  float* out_y      = (float*)d_out;                  // [64,40] = 2560
  float* out_latent = (float*)d_out + (size_t)G * 40; // [64,32] = 2048
  k_mlp1<<<(G * 32 + 255) / 256, 256, 0, stream>>>(pf, W1, b1, lat, out_latent);
  k_mlp2<<<(G * 128 + 255) / 256, 256, 0, stream>>>(lat, W2, b2, ymid);
  k_mlp3<<<(G * 40 + 255) / 256, 256, 0, stream>>>(ymid, W3, b3, out_y);
}
